// HyperbolicMambaLorentz_52020643889828
// MI455X (gfx1250) — compile-verified, matrix operand order checked by reading, not run on hardware
//
#include <hip/hip_runtime.h>

typedef __attribute__((ext_vector_type(16))) __bf16 v16bf;
typedef __attribute__((ext_vector_type(8)))  float  v8f;

#define B_DIM 8192
#define D_DIM 4096
#define H_DIM 16384
#define EPSF  1e-6f
#define MAXTN 10.0f

// LDS row stride in bf16 elements (40 bf16 = 80 bytes = 5*16B, keeps 16B alignment)
#define LDS_STRIDE 40

// ---------------------------------------------------------------------------
// Block-wide sum reduction (8 waves of 32 lanes). Returns full sum to every
// thread. Caller provides __shared__ float sred[8].
// ---------------------------------------------------------------------------
__device__ inline float block_reduce(float v, float* sred) {
    #pragma unroll
    for (int o = 16; o > 0; o >>= 1) v += __shfl_down(v, o, 32);
    const int lane = threadIdx.x & 31;
    const int w    = threadIdx.x >> 5;
    __syncthreads();                 // protect sred reuse across calls
    if (lane == 0) sred[w] = v;
    __syncthreads();
    float s = 0.0f;
    #pragma unroll
    for (int i = 0; i < 8; ++i) s += sred[i];
    return s;
}

// ---------------------------------------------------------------------------
// Load a 16x32 bf16 WMMA A/B fragment from LDS (row-major, padded rows).
// Per CDNA5 ISA 16-bit A-matrix layout: lane L -> row = L%16, h = L/16;
// VGPRs 0-3 hold K = 8h..8h+7 (16 contiguous bytes), VGPRs 4-7 hold
// K = 16+8h..16+8h+7. Two ds_load_b128 per lane.
// ---------------------------------------------------------------------------
__device__ inline v16bf load_frag(const __bf16* base, int row0, int lane) {
    const int m = lane & 15, h = lane >> 4;
    const char* rp = (const char*)(base + (size_t)(row0 + m) * LDS_STRIDE);
    union { uint4 u[2]; v16bf v; } un;
    un.u[0] = *(const uint4*)(rp + 16 * h);
    un.u[1] = *(const uint4*)(rp + 32 + 16 * h);
    return un.v;
}

// ---------------------------------------------------------------------------
// Tangent: per-row logmap0.  out[j] = (j==0) ? 0 : acosh(x0) * z[j] / |y|
// ---------------------------------------------------------------------------
__global__ __launch_bounds__(256)
void tangent_kernel(const float* __restrict__ z, __bf16* __restrict__ out) {
    __shared__ float red[8];
    const size_t row = blockIdx.x;
    const float* zr = z + row * D_DIM;
    float vals[16];
    float part = 0.0f;
    #pragma unroll
    for (int i = 0; i < 16; ++i) {
        const int j = threadIdx.x + 256 * i;
        const float v = zr[j];
        vals[i] = v;
        if (j != 0) part += v * v;
    }
    const float ss    = block_reduce(part, red);
    const float x0    = fmaxf(zr[0], 1.0f + EPSF);
    const float ynorm = sqrtf(fmaxf(ss, EPSF));
    const float d     = logf(x0 + sqrtf(x0 * x0 - 1.0f));   // acosh
    const float scale = d / ynorm;
    __bf16* orow = out + row * D_DIM;
    #pragma unroll
    for (int i = 0; i < 16; ++i) {
        const int j = threadIdx.x + 256 * i;
        orow[j] = (j == 0) ? (__bf16)0.0f : (__bf16)(vals[i] * scale);
    }
}

// ---------------------------------------------------------------------------
// Tiled WMMA GEMM:  C = epilogue(A_bf16[MxK] @ bf16(B_f32[KxN]) + bias[N])
//   EPI==0 : relu -> bf16     EPI==1 : identity -> f32
// Block tile 128x128, K-step 32, 8 waves (2x4), 64x32 per wave.
// Double-buffered LDS; A staged via global_load_async_to_lds_b128 (ASYNCcnt),
// B staged through VGPRs (fp32->bf16 convert) with immediate-offset loads.
// All LDS accesses keep a single __shared__ pointer root + integer offsets so
// they lower to ds_* (not flat_*).
// ---------------------------------------------------------------------------
template <int EPI, int N, int K>
__global__ __launch_bounds__(256)
void gemm_wmma_bf16(const __bf16* __restrict__ A, const float* __restrict__ Bsrc,
                    const float* __restrict__ bias, void* __restrict__ Cout) {
    __shared__ __align__(16) __bf16 As[2][128 * LDS_STRIDE];
    __shared__ __align__(16) __bf16 Bs[2][128 * LDS_STRIDE];
    constexpr unsigned BUF_BYTES = 128 * LDS_STRIDE * 2;   // 10240

    const int tid  = threadIdx.x;
    const int lane = tid & 31;
    const int wave = tid >> 5;
    const int wm   = wave >> 2;          // 0..1 (M)
    const int wn   = wave & 3;           // 0..3 (N)
    const int block_n = blockIdx.x * 128;
    const int block_m = blockIdx.y * 128;

    v8f acc[4][2] = {};

    const int a_row = tid >> 2;          // 0..63 (+64 for second copy)
    const int a_cg  = tid & 3;
    const int b_n   = tid & 127;
    const int b_kh  = tid >> 7;          // 0/1
    constexpr int kTiles = K >> 5;

    // --- running global pointers (advance by 32 k per tile)
    const __bf16* ag = A + (size_t)(block_m + a_row) * K + 8 * a_cg;
    const float*  bg = Bsrc + (size_t)(16 * b_kh) * N + block_n + b_n;

    // --- LDS byte offsets (async A dest uses low 32 bits of flat address)
    const uint32_t aL0  = (uint32_t)(uintptr_t)&As[0][a_row * LDS_STRIDE + 8 * a_cg];
    const unsigned bOff = (unsigned)(b_n * (LDS_STRIDE * 2) + 32 * b_kh);

    auto stageA = [&](const __bf16* g, int s) {
        unsigned long long p0 = (unsigned long long)(uintptr_t)g;
        unsigned long long p1 = (unsigned long long)(uintptr_t)(g + (size_t)64 * K);
        uint32_t l0 = aL0 + (unsigned)s * BUF_BYTES;
        uint32_t l1 = l0 + 64 * (LDS_STRIDE * 2);
        asm volatile("global_load_async_to_lds_b128 %0, %1, off"
                     :: "v"(l0), "v"(p0) : "memory");
        asm volatile("global_load_async_to_lds_b128 %0, %1, off"
                     :: "v"(l1), "v"(p1) : "memory");
    };
    auto loadB = [&](const float* p, float* fr) {
        #pragma unroll
        for (int kk = 0; kk < 16; ++kk) fr[kk] = p[(size_t)kk * N];  // imm offsets
    };
    // Single pointer root (Bs) + integer offset -> lowers to ds_store_b128.
    auto storeB = [&](const float* fr, int s) {
        __align__(16) __bf16 tmp[16];
        #pragma unroll
        for (int kk = 0; kk < 16; ++kk) tmp[kk] = (__bf16)fr[kk];
        uint4* dst = (uint4*)((char*)Bs + (unsigned)s * BUF_BYTES + bOff);
        dst[0] = ((const uint4*)tmp)[0];
        dst[1] = ((const uint4*)tmp)[1];
    };

    // ---- prologue: stage tile 0 into buffer 0
    {
        float br0[16];
        stageA(ag, 0);
        loadB(bg, br0);
        storeB(br0, 0);
    }

    int cur = 0;
    const __bf16* agNext = ag + 32;
    const float*  bgNext = bg + (size_t)32 * N;

    for (int kt = 0; kt < kTiles; ++kt) {
        const bool more = (kt + 1 < kTiles);
        const int  nxt  = cur ^ 1;

        // next B tile: global -> regs (issued before the barrier; overlaps)
        float brn[16];
        if (more) loadB(bgNext, brn);

        // my async copies into `cur` (issued last iteration) must be in LDS
        // before I signal; barrier then makes every wave's writes visible.
        asm volatile("s_wait_asynccnt 0" ::: "memory");
        __syncthreads();

        // next A tile: async global->LDS into `nxt`. Safe: every wave finished
        // reading `nxt` before the barrier we just passed (per-wave dscnt wait).
        if (more) stageA(agNext, nxt);

        // fragments from `cur` (single root -> ds_load_b128)
        v16bf af[4], bfr[2];
        #pragma unroll
        for (int tm = 0; tm < 4; ++tm)
            af[tm]  = load_frag((const __bf16*)((char*)As + (unsigned)cur * BUF_BYTES),
                                wm * 64 + tm * 16, lane);
        #pragma unroll
        for (int tn = 0; tn < 2; ++tn)
            bfr[tn] = load_frag((const __bf16*)((char*)Bs + (unsigned)cur * BUF_BYTES),
                                wn * 32 + tn * 16, lane);

        // convert + store next B tile into `nxt`
        if (more) storeB(brn, nxt);

        #pragma unroll
        for (int tm = 0; tm < 4; ++tm)
            #pragma unroll
            for (int tn = 0; tn < 2; ++tn)
                acc[tm][tn] = __builtin_amdgcn_wmma_f32_16x16x32_bf16(
                    false, af[tm], false, bfr[tn], (short)0, acc[tm][tn], false, false);

        if (more) { agNext += 32; bgNext += (size_t)32 * N; }
        cur = nxt;
    }

    // ---- epilogue: C/D layout: lane L, reg r -> row = 8*(L/16)+r, col = L%16
    const int ln = lane & 15, lh = lane >> 4;
    #pragma unroll
    for (int tn = 0; tn < 2; ++tn) {
        const int col = block_n + wn * 32 + tn * 16 + ln;
        const float bv = bias[col];
        #pragma unroll
        for (int tm = 0; tm < 4; ++tm) {
            #pragma unroll
            for (int r = 0; r < 8; ++r) {
                const int rowg = block_m + wm * 64 + tm * 16 + lh * 8 + r;
                const float v = acc[tm][tn][r] + bv;
                if (EPI == 0)
                    ((__bf16*)Cout)[(size_t)rowg * N + col] = (__bf16)fmaxf(v, 0.0f);
                else
                    ((float*)Cout)[(size_t)rowg * N + col] = v;
            }
        }
    }
}

// ---------------------------------------------------------------------------
// Final fused epilogue per row: sigmoid gate, Lorentz inner product, project,
// expmap0, tanh.
// ---------------------------------------------------------------------------
__global__ __launch_bounds__(256)
void final_kernel(const float* __restrict__ z_t, const float* __restrict__ v_raw,
                  const float* __restrict__ glog, float* __restrict__ z_next,
                  float* __restrict__ v_proj_out) {
    __shared__ float red[8];
    const size_t row = blockIdx.x;
    const float* zr = z_t  + row * D_DIM;
    const float* vr = v_raw + row * D_DIM;
    const float* gr = glog + row * D_DIM;

    float zv[16], vp[16];
    float pl = 0.0f;
    #pragma unroll
    for (int i = 0; i < 16; ++i) {
        const int j = threadIdx.x + 256 * i;
        const float z = zr[j];
        const float g = gr[j];
        const float v = vr[j];
        const float gate = 1.0f / (1.0f + expf(-g));
        const float p = gate * v;
        zv[i] = z; vp[i] = p;
        pl += (j == 0) ? -z * p : z * p;
    }
    const float ldot = block_reduce(pl, red);

    float ps = 0.0f;
    #pragma unroll
    for (int i = 0; i < 16; ++i) {
        const int j = threadIdx.x + 256 * i;
        const float vpj = vp[i] + ldot * zv[i];
        vp[i] = vpj;
        ps += (j == 0) ? -vpj * vpj : vpj * vpj;
    }
    const float sq = block_reduce(ps, red);

    const float n  = fminf(fmaxf(sqrtf(fmaxf(sq, EPSF)), EPSF), MAXTN);
    const float ch = coshf(n);
    const float sh = sinhf(n) / n;

    float* zo = z_next     + row * D_DIM;
    float* vo = v_proj_out + row * D_DIM;
    #pragma unroll
    for (int i = 0; i < 16; ++i) {
        const int j = threadIdx.x + 256 * i;
        const float e = sh * vp[i] + ((j == 0) ? ch : 0.0f);
        zo[j] = tanhf(e);
        vo[j] = vp[i];
    }
}

// ---------------------------------------------------------------------------
// Workspace layout: tangent bf16 (64MiB) | h bf16 (256MiB) | v_raw f32
// (128MiB) | glogit f32 (128MiB)
// ---------------------------------------------------------------------------
extern "C" void kernel_launch(void* const* d_in, const int* in_sizes, int n_in,
                              void* d_out, int out_size, void* d_ws, size_t ws_size,
                              hipStream_t stream) {
    const float* z_t = (const float*)d_in[0];
    const float* W1  = (const float*)d_in[1];
    const float* b1  = (const float*)d_in[2];
    const float* W2  = (const float*)d_in[3];
    const float* b2  = (const float*)d_in[4];
    const float* Wg  = (const float*)d_in[5];
    const float* bg  = (const float*)d_in[6];

    const size_t TAN_BYTES = (size_t)B_DIM * D_DIM * sizeof(__bf16);
    const size_t H_BYTES   = (size_t)B_DIM * H_DIM * sizeof(__bf16);
    const size_t VR_BYTES  = (size_t)B_DIM * D_DIM * sizeof(float);

    char* ws = (char*)d_ws;
    __bf16* tan_bf = (__bf16*)(ws);
    __bf16* h_bf   = (__bf16*)(ws + TAN_BYTES);
    float*  v_raw  = (float*)(ws + TAN_BYTES + H_BYTES);
    float*  glogit = (float*)(ws + TAN_BYTES + H_BYTES + VR_BYTES);

    float* out_znext = (float*)d_out;
    float* out_vproj = out_znext + (size_t)B_DIM * D_DIM;

    tangent_kernel<<<B_DIM, 256, 0, stream>>>(z_t, tan_bf);

    gemm_wmma_bf16<0, H_DIM, D_DIM><<<dim3(H_DIM / 128, B_DIM / 128), 256, 0, stream>>>(
        tan_bf, W1, b1, (void*)h_bf);

    gemm_wmma_bf16<1, D_DIM, H_DIM><<<dim3(D_DIM / 128, B_DIM / 128), 256, 0, stream>>>(
        h_bf, W2, b2, (void*)v_raw);

    gemm_wmma_bf16<1, D_DIM, D_DIM><<<dim3(D_DIM / 128, B_DIM / 128), 256, 0, stream>>>(
        tan_bf, Wg, bg, (void*)glogit);

    final_kernel<<<B_DIM, 256, 0, stream>>>(z_t, v_raw, glogit, out_znext, out_vproj);
}
